// Recurrent_87411174408879
// MI455X (gfx1250) — compile-verified, matrix operand order checked by reading, not run on hardware
//
#include <hip/hip_runtime.h>
#include <hip/hip_bf16.h>

typedef __attribute__((ext_vector_type(16))) _Float16 v16h;
typedef __attribute__((ext_vector_type(8)))  _Float16 v8h;
typedef __attribute__((ext_vector_type(8)))  float    v8f;

#define NEG_SLOPE 0.01f

__device__ __forceinline__ v8f wmma_f16(v16h a, v16h b, v8f c) {
  // D = A(16x32 f16) * B(32x16 f16) + C(16x16 f32)
  return __builtin_amdgcn_wmma_f32_16x16x32_f16(
      /*neg_a=*/false, a, /*neg_b=*/false, b,
      /*c_mod=*/(short)0, c, /*reuse_a=*/false, /*reuse_b=*/false);
}

// A fragment (16x32 f16) from LDS, row-major buf with strideH halfs per row.
// lane<16: M=lane, K-halfs {0..7,16..23}; lane>=16: M=lane-16, K {8..15,24..31}.
__device__ __forceinline__ v16h load_fragA(const _Float16* buf, int strideH,
                                           int k0, int lane) {
  int m  = lane & 15;
  int bk = (lane < 16) ? 0 : 8;
  const _Float16* p = buf + m * strideH + k0 + bk;
  v8h lo = *(const v8h*)(p);       // K = bk .. bk+7      (ds_load_b128)
  v8h hi = *(const v8h*)(p + 16);  // K = bk+16 .. bk+23  (ds_load_b128)
  v16h r;
#pragma unroll
  for (int i = 0; i < 8; ++i) { r[i] = lo[i]; r[i + 8] = hi[i]; }
  return r;
}

// B fragment (32x16 f16) gathered from f32 weight matrix W[K][N] (row-major),
// tile at (k0, n0); zero-padded outside [K, N].
__device__ __forceinline__ v16h load_fragB(const float* W, int K, int N,
                                           int k0, int n0, int lane) {
  int n  = (lane & 15) + n0;
  int bk = (lane < 16) ? 0 : 8;
  v16h r;
#pragma unroll
  for (int j = 0; j < 16; ++j) {
    int k = k0 + bk + ((j < 8) ? j : (j + 8));
    float v = (k < K && n < N) ? W[k * N + n] : 0.0f;
    r[j] = (_Float16)v;
  }
  return r;
}

// Apply bias + activation to a 16x16 f32 accumulator tile and scatter the
// real columns (col < outN) as f16 into the next layer's LDS input buffer.
__device__ __forceinline__ void store_act_f16(_Float16* buf, int strideH,
                                              int outN, int n0, v8f c,
                                              float bias, int lane, bool sig) {
  int n   = lane & 15;
  int col = n0 + n;
  int mb  = (lane < 16) ? 0 : 8;
  if (col < outN) {
#pragma unroll
    for (int r = 0; r < 8; ++r) {
      float z = c[r] + bias;
      float a = sig ? (1.0f / (1.0f + __expf(-z)))
                    : ((z > 0.0f) ? z : NEG_SLOPE * z);
      buf[(mb + r) * strideH + col] = (_Float16)a;
    }
  }
}

__global__ void __launch_bounds__(128)
recurrent_wmma_kernel(const float* __restrict__ w,
                      const float* Wh1, const float* bh1,
                      const float* Wh2, const float* bh2,
                      const float* Wh3, const float* bh3,
                      const float* Wx1, const float* bx1,
                      const float* Wx2, const float* bx2,
                      const float* Wx3, const float* bx3,
                      const int* __restrict__ num_points,
                      float* __restrict__ out, int Brows) {
  // per-wave LDS slices: hbuf 16x32, buf50 16x64, buf20 16x32 (f16) + 32 f32
  __shared__ __align__(16) _Float16 s_act[4][2048];
  __shared__ __align__(16) float s_x[4][32];

  const int lane = threadIdx.x & 31;
  const int wv   = threadIdx.x >> 5;
  const int r0   = (blockIdx.x * 4 + wv) * 16;   // first batch row of this wave
  if (r0 >= Brows) return;                       // wave-uniform
  const int T = *num_points;

  _Float16* bufh  = s_act[wv];         // 16 x 32  (h carry, K-pad 32)
  _Float16* buf50 = s_act[wv] + 512;   // 16 x 64  (50-wide act, K-pad 64)
  _Float16* buf20 = s_act[wv] + 1536;  // 16 x 32  (20-wide act, K-pad 32)
  float*    bufx  = s_x[wv];

  // ---- weights resident in VGPRs as WMMA B fragments ----
  v16h Bh1[4], Bx1[4], Bh2[2][2], Bx2[2][2], Bh3, Bx3;
#pragma unroll
  for (int j = 0; j < 4; ++j) {
    Bh1[j] = load_fragB(Wh1, 2, 50, 0, 16 * j, lane);
    Bx1[j] = load_fragB(Wx1, 2, 50, 0, 16 * j, lane);
  }
#pragma unroll
  for (int j = 0; j < 2; ++j)
#pragma unroll
    for (int kk = 0; kk < 2; ++kk) {
      Bh2[j][kk] = load_fragB(Wh2, 50, 20, 32 * kk, 16 * j, lane);
      Bx2[j][kk] = load_fragB(Wx2, 50, 20, 32 * kk, 16 * j, lane);
    }
  Bh3 = load_fragB(Wh3, 20, 2, 0, 0, lane);
  Bx3 = load_fragB(Wx3, 20, 2, 0, 0, lane);

  // ---- biases: one value per lane per N-tile ----
  const int n = lane & 15;
  float vbh1[4], vbx1[4], vbh2[2], vbx2[2];
#pragma unroll
  for (int j = 0; j < 4; ++j) {
    int c = 16 * j + n;
    vbh1[j] = (c < 50) ? bh1[c] : 0.0f;
    vbx1[j] = (c < 50) ? bx1[c] : 0.0f;
  }
#pragma unroll
  for (int j = 0; j < 2; ++j) {
    int c = 16 * j + n;
    vbh2[j] = (c < 20) ? bh2[c] : 0.0f;
    vbx2[j] = (c < 20) ? bx2[c] : 0.0f;
  }
  float vbh3 = (n < 2) ? bh3[n] : 0.0f;
  float vbx3 = (n < 2) ? bx3[n] : 0.0f;

  // ---- zero this wave's LDS slice once (pads stay zero forever) ----
  {
    unsigned int* z = (unsigned int*)s_act[wv];
    for (int i = lane; i < 1024; i += 32) z[i] = 0u;
  }
  // initial h = w[r0 .. r0+15][0..1]
  if (lane < 16) {
    int rr = (r0 + lane < Brows) ? (r0 + lane) : (Brows - 1);
    bufh[lane * 32 + 0] = (_Float16)w[(size_t)rr * 2 + 0];
    bufh[lane * 32 + 1] = (_Float16)w[(size_t)rr * 2 + 1];
  }
  // single wave owns its LDS slice; per-wave LDS ops are in-order -> no barrier

  const size_t outRowStride = (size_t)T * 2;
  const int outRow = r0 + (lane >> 1);
  const int outCol = lane & 1;

  for (int t = 0; t < T; ++t) {
    // ===== h path =====
    {  // 2 -> 50, lrelu
      v16h a = load_fragA(bufh, 32, 0, lane);
      v8f acc[4];
#pragma unroll
      for (int j = 0; j < 4; ++j) { v8f c0 = {}; acc[j] = wmma_f16(a, Bh1[j], c0); }
#pragma unroll
      for (int j = 0; j < 4; ++j)
        store_act_f16(buf50, 64, 50, 16 * j, acc[j], vbh1[j], lane, false);
    }
    {  // 50 -> 20, lrelu
      v16h a0 = load_fragA(buf50, 64, 0, lane);
      v16h a1 = load_fragA(buf50, 64, 32, lane);
#pragma unroll
      for (int j = 0; j < 2; ++j) {
        v8f c0 = {};
        c0 = wmma_f16(a0, Bh2[j][0], c0);
        c0 = wmma_f16(a1, Bh2[j][1], c0);
        store_act_f16(buf20, 32, 20, 16 * j, c0, vbh2[j], lane, false);
      }
    }
    {  // 20 -> 2, lrelu : new h carry
      v16h a = load_fragA(buf20, 32, 0, lane);
      v8f c0 = {};
      c0 = wmma_f16(a, Bh3, c0);
      store_act_f16(bufh, 32, 2, 0, c0, vbh3, lane, false);
    }
    // ===== x path =====
    {  // 2 -> 50, sigmoid (reads the just-written new h)
      v16h a = load_fragA(bufh, 32, 0, lane);
      v8f acc[4];
#pragma unroll
      for (int j = 0; j < 4; ++j) { v8f c0 = {}; acc[j] = wmma_f16(a, Bx1[j], c0); }
#pragma unroll
      for (int j = 0; j < 4; ++j)
        store_act_f16(buf50, 64, 50, 16 * j, acc[j], vbx1[j], lane, true);
    }
    {  // 50 -> 20, sigmoid
      v16h a0 = load_fragA(buf50, 64, 0, lane);
      v16h a1 = load_fragA(buf50, 64, 32, lane);
#pragma unroll
      for (int j = 0; j < 2; ++j) {
        v8f c0 = {};
        c0 = wmma_f16(a0, Bx2[j][0], c0);
        c0 = wmma_f16(a1, Bx2[j][1], c0);
        store_act_f16(buf20, 32, 20, 16 * j, c0, vbx2[j], lane, true);
      }
    }
    {  // 20 -> 2, sigmoid : emit x_t
      v16h a = load_fragA(buf20, 32, 0, lane);
      v8f c0 = {};
      c0 = wmma_f16(a, Bx3, c0);
      int mb = (lane < 16) ? 0 : 8;
      if (n < 2) {
#pragma unroll
        for (int r = 0; r < 8; ++r) {
          float z = c0[r] + vbx3;
          bufx[(mb + r) * 2 + n] = 1.0f / (1.0f + __expf(-z));
        }
      }
      // all 32 lanes cooperatively store [16 rows x 2 cols] f32 for step t
      float xv = bufx[lane];
      if (outRow < Brows)
        out[(size_t)outRow * outRowStride + (size_t)t * 2 + outCol] = xv;
    }
  }
}

extern "C" void kernel_launch(void* const* d_in, const int* in_sizes, int n_in,
                              void* d_out, int out_size, void* d_ws, size_t ws_size,
                              hipStream_t stream) {
  const float* w   = (const float*)d_in[0];
  const float* Wh1 = (const float*)d_in[1];
  const float* bh1 = (const float*)d_in[2];
  const float* Wh2 = (const float*)d_in[3];
  const float* bh2 = (const float*)d_in[4];
  const float* Wh3 = (const float*)d_in[5];
  const float* bh3 = (const float*)d_in[6];
  const float* Wx1 = (const float*)d_in[7];
  const float* bx1 = (const float*)d_in[8];
  const float* Wx2 = (const float*)d_in[9];
  const float* bx2 = (const float*)d_in[10];
  const float* Wx3 = (const float*)d_in[11];
  const float* bx3 = (const float*)d_in[12];
  const int* num_points = (const int*)d_in[13];

  int Brows  = in_sizes[0] / 2;          // 65536
  int blocks = (Brows + 63) / 64;        // 4 waves x 16 rows per block
  recurrent_wmma_kernel<<<blocks, 128, 0, stream>>>(
      w, Wh1, bh1, Wh2, bh2, Wh3, bh3,
      Wx1, bx1, Wx2, bx2, Wx3, bx3,
      num_points, (float*)d_out, Brows);
}
